// PairAttentionLayer_85134841741412
// MI455X (gfx1250) — matrix operand order repaired and checked by reading, not compiled, not run on hardware
//
#include <hip/hip_runtime.h>
#include <hip/hip_bf16.h>

typedef __bf16 v16bf __attribute__((ext_vector_type(16)));
typedef float  v8f   __attribute__((ext_vector_type(8)));

#define NEG_INF (-1e12f)
#define DK      128     // Dk = Dq1 = Dq2 = 128
#define A_DIM   75
#define A_PAD   80      // 5 WMMA N-tiles of 16
#define ROWS_PER_WAVE   32
#define ROWS_PER_BLOCK  128   // 4 waves * 32 rows
#define NSPLIT  8       // L-splits for the context pass

// gfx1250 has a hardware tanh transcendental (V_TANH_F32); use it if exposed.
#if defined(__has_builtin)
#  if __has_builtin(__builtin_amdgcn_tanhf)
#    define TANHF(x) __builtin_amdgcn_tanhf(x)
#  elif __has_builtin(__builtin_amdgcn_tanh_f32)
#    define TANHF(x) __builtin_amdgcn_tanh_f32(x)
#  endif
#endif
#ifndef TANHF
#  define TANHF(x) tanhf(x)
#endif

// ---------------------------------------------------------------------------
// Kernel 1: scores[b][l] = mask ? Ws . tanh(K[l,b,:] Wk^T + qterm) : -1e12
// One block = 4 waves; each wave computes 32 rows of L (two 16-row M-tiles)
// so every B-fragment ds_load feeds two v_wmma_f32_16x16x32_bf16.
// ---------------------------------------------------------------------------
__global__ __launch_bounds__(128)
void pairattn_scores_kernel(const float* __restrict__ keys,
                            const float* __restrict__ q1,
                            const float* __restrict__ q2,
                            const int*   __restrict__ key_mask,
                            const float* __restrict__ Wk,
                            const float* __restrict__ Wq1,
                            const float* __restrict__ Wq2,
                            const float* __restrict__ Ws,
                            float* __restrict__ scores,
                            int L, int B)
{
    __shared__ __bf16 wk_lds[A_PAD * DK];   // Wk as bf16, zero-padded rows
    __shared__ float  qterm_lds[A_PAD];     // Wq1.q1[b] + Wq2.q2[b]
    __shared__ float  ws_lds[A_PAD];        // Ws, zero-padded

    const int tid = threadIdx.x;
    const int b   = blockIdx.y;

    // Stage Wk (f32 -> bf16) into LDS.
    for (int idx = tid; idx < A_PAD * DK; idx += 128) {
        const int a = idx >> 7;
        const int d = idx & (DK - 1);
        const float v = (a < A_DIM) ? Wk[(size_t)a * DK + d] : 0.0f;
        wk_lds[idx] = (__bf16)v;
    }
    // Per-batch query bias term and Ws.
    if (tid < A_PAD) {
        float s = 0.0f;
        if (tid < A_DIM) {
            const float* q1b = q1  + (size_t)b   * DK;
            const float* q2b = q2  + (size_t)b   * DK;
            const float* w1  = Wq1 + (size_t)tid * DK;
            const float* w2  = Wq2 + (size_t)tid * DK;
            #pragma unroll 4
            for (int d = 0; d < DK; ++d) s += w1[d] * q1b[d] + w2[d] * q2b[d];
            ws_lds[tid] = Ws[tid];
        } else {
            ws_lds[tid] = 0.0f;
        }
        qterm_lds[tid] = s;
    }
    __syncthreads();

    const int wave   = tid >> 5;
    const int lane   = tid & 31;
    const int hlf    = lane >> 4;     // lane half (0: lanes 0-15, 1: lanes 16-31)
    const int mcol   = lane & 15;
    const int m_base = blockIdx.x * ROWS_PER_BLOCK + wave * ROWS_PER_WAVE;

    int row0 = m_base + mcol;
    int row1 = m_base + 16 + mcol;
    if (row0 >= L) row0 = L - 1;      // clamp; dup rows discarded at write
    if (row1 >= L) row1 = L - 1;
    const float* rp0 = keys + (size_t)row0 * B * DK + (size_t)b * DK;
    const float* rp1 = keys + (size_t)row1 * B * DK + (size_t)b * DK;

    v8f acc0[5], acc1[5];
    #pragma unroll
    for (int t = 0; t < 5; ++t)
        #pragma unroll
        for (int v = 0; v < 8; ++v) { acc0[t][v] = 0.0f; acc1[t][v] = 0.0f; }

    #pragma unroll
    for (int kk = 0; kk < 4; ++kk) {
        // A-fragments (16x32 bf16): lane holds row M=lane%16.
        // e<8 -> K = kk*32 + hlf*8 + e ; e>=8 -> K = kk*32 + 16 + hlf*8 + (e-8)
        const int c0 = kk * 32 + hlf * 8;
        v16bf af0, af1;
        {
            const float4 x0 = *(const float4*)(rp0 + c0);
            const float4 x1 = *(const float4*)(rp0 + c0 + 4);
            const float4 x2 = *(const float4*)(rp0 + c0 + 16);
            const float4 x3 = *(const float4*)(rp0 + c0 + 20);
            af0[0]  = (__bf16)x0.x; af0[1]  = (__bf16)x0.y; af0[2]  = (__bf16)x0.z; af0[3]  = (__bf16)x0.w;
            af0[4]  = (__bf16)x1.x; af0[5]  = (__bf16)x1.y; af0[6]  = (__bf16)x1.z; af0[7]  = (__bf16)x1.w;
            af0[8]  = (__bf16)x2.x; af0[9]  = (__bf16)x2.y; af0[10] = (__bf16)x2.z; af0[11] = (__bf16)x2.w;
            af0[12] = (__bf16)x3.x; af0[13] = (__bf16)x3.y; af0[14] = (__bf16)x3.z; af0[15] = (__bf16)x3.w;
        }
        {
            const float4 x0 = *(const float4*)(rp1 + c0);
            const float4 x1 = *(const float4*)(rp1 + c0 + 4);
            const float4 x2 = *(const float4*)(rp1 + c0 + 16);
            const float4 x3 = *(const float4*)(rp1 + c0 + 20);
            af1[0]  = (__bf16)x0.x; af1[1]  = (__bf16)x0.y; af1[2]  = (__bf16)x0.z; af1[3]  = (__bf16)x0.w;
            af1[4]  = (__bf16)x1.x; af1[5]  = (__bf16)x1.y; af1[6]  = (__bf16)x1.z; af1[7]  = (__bf16)x1.w;
            af1[8]  = (__bf16)x2.x; af1[9]  = (__bf16)x2.y; af1[10] = (__bf16)x2.z; af1[11] = (__bf16)x2.w;
            af1[12] = (__bf16)x3.x; af1[13] = (__bf16)x3.y; af1[14] = (__bf16)x3.z; af1[15] = (__bf16)x3.w;
        }

        #pragma unroll
        for (int t = 0; t < 5; ++t) {
            // B-fragment (32x16 bf16): lane holds column N=lane%16,
            // e<8 -> K = kk*32 + hlf*8 + e ; e>=8 -> K = kk*32 + 16 + hlf*8 + (e-8).
            const __bf16* wp = wk_lds + (t * 16 + mcol) * DK + kk * 32 + hlf * 8;
            v16bf bf;
            #pragma unroll
            for (int e = 0; e < 16; ++e) bf[e] = wp[(e & 7) + ((e >> 3) << 4)];
            acc0[t] = __builtin_amdgcn_wmma_f32_16x16x32_bf16(
                false, af0, false, bf, (short)0, acc0[t], false, false);
            acc1[t] = __builtin_amdgcn_wmma_f32_16x16x32_bf16(
                false, af1, false, bf, (short)0, acc1[t], false, false);
        }
    }

    // Epilogue: score_row = sum_a Ws[a] * tanh(acc + qterm[a]).
    // C layout: VGPR v, lanes 0-15 -> M=v ; lanes 16-31 -> M=v+8 ; N=lane%16.
    float p0[8], p1[8];
    #pragma unroll
    for (int v = 0; v < 8; ++v) { p0[v] = 0.0f; p1[v] = 0.0f; }
    #pragma unroll
    for (int t = 0; t < 5; ++t) {
        const int a = t * 16 + mcol;
        const float w  = ws_lds[a];
        const float qt = qterm_lds[a];
        #pragma unroll
        for (int v = 0; v < 8; ++v) {
            p0[v] += w * TANHF(acc0[t][v] + qt);
            p1[v] += w * TANHF(acc1[t][v] + qt);
        }
    }
    // Butterfly-reduce across each 16-lane half (masks 1,2,4,8 stay in-half).
    #pragma unroll
    for (int v = 0; v < 8; ++v) {
        float a0 = p0[v], a1 = p1[v];
        a0 += __shfl_xor(a0, 1, 32);  a1 += __shfl_xor(a1, 1, 32);
        a0 += __shfl_xor(a0, 2, 32);  a1 += __shfl_xor(a1, 2, 32);
        a0 += __shfl_xor(a0, 4, 32);  a1 += __shfl_xor(a1, 4, 32);
        a0 += __shfl_xor(a0, 8, 32);  a1 += __shfl_xor(a1, 8, 32);
        p0[v] = a0; p1[v] = a1;
    }
    if (mcol == 0) {
        #pragma unroll
        for (int v = 0; v < 8; ++v) {
            const int r0 = m_base + hlf * 8 + v;
            const int r1 = r0 + 16;
            if (r0 < L) {
                scores[(size_t)b * L + r0] =
                    (key_mask[(size_t)r0 * B + b] != 1) ? NEG_INF : p0[v];
            }
            if (r1 < L) {
                scores[(size_t)b * L + r1] =
                    (key_mask[(size_t)r1 * B + b] != 1) ? NEG_INF : p1[v];
            }
        }
    }
}

// ---------------------------------------------------------------------------
// Kernel 2a: per-batch softmax, normalized in place: scores -> probs.
// ---------------------------------------------------------------------------
__global__ __launch_bounds__(256)
void pairattn_softmax_kernel(float* __restrict__ scores, int L, int B)
{
    __shared__ float probs[4096];
    __shared__ float red[8];

    const int b   = blockIdx.x;
    const int tid = threadIdx.x;

    float m = NEG_INF;
    for (int l = tid; l < L; l += 256)
        m = fmaxf(m, scores[(size_t)b * L + l]);
    #pragma unroll
    for (int off = 16; off; off >>= 1) m = fmaxf(m, __shfl_xor(m, off, 32));
    if ((tid & 31) == 0) red[tid >> 5] = m;
    __syncthreads();
    m = red[0];
    #pragma unroll
    for (int w = 1; w < 8; ++w) m = fmaxf(m, red[w]);
    __syncthreads();

    float s = 0.0f;
    for (int l = tid; l < L; l += 256) {
        const float e = __expf(scores[(size_t)b * L + l] - m);
        probs[l] = e;
        s += e;
    }
    #pragma unroll
    for (int off = 16; off; off >>= 1) s += __shfl_xor(s, off, 32);
    __syncthreads();
    if ((tid & 31) == 0) red[tid >> 5] = s;
    __syncthreads();
    s = 0.0f;
    #pragma unroll
    for (int w = 0; w < 8; ++w) s += red[w];
    const float inv = 1.0f / s;

    for (int l = tid; l < L; l += 256)
        scores[(size_t)b * L + l] = probs[l] * inv;
}

// ---------------------------------------------------------------------------
// Kernel 2b: partial context sums over NSPLIT chunks of L (keys L2-resident).
// ---------------------------------------------------------------------------
__global__ __launch_bounds__(256)
void pairattn_ctx_partial_kernel(const float* __restrict__ keys,
                                 const float* __restrict__ probs,
                                 float* __restrict__ partials,
                                 int L, int B)
{
    __shared__ float hbuf[DK];
    const int b   = blockIdx.x;
    const int sp  = blockIdx.y;
    const int tid = threadIdx.x;
    const int d   = tid & (DK - 1);
    const int lh  = tid >> 7;

    const int chunk = (L + NSPLIT - 1) / NSPLIT;
    const int l0 = sp * chunk;
    int l1 = l0 + chunk; if (l1 > L) l1 = L;

    float acc = 0.0f;
    for (int l = l0 + lh; l < l1; l += 2)
        acc += probs[(size_t)b * L + l] *
               keys[(size_t)l * B * DK + (size_t)b * DK + d];
    if (lh == 1) hbuf[d] = acc;
    __syncthreads();
    if (lh == 0)
        partials[((size_t)b * NSPLIT + sp) * DK + d] = acc + hbuf[d];
}

// ---------------------------------------------------------------------------
// Kernel 2c: reduce NSPLIT partials -> context (1, B, Dk).
// ---------------------------------------------------------------------------
__global__ __launch_bounds__(128)
void pairattn_ctx_reduce_kernel(const float* __restrict__ partials,
                                float* __restrict__ out, int B)
{
    const int b = blockIdx.x;
    const int d = threadIdx.x;
    float s = 0.0f;
    #pragma unroll
    for (int sp = 0; sp < NSPLIT; ++sp)
        s += partials[((size_t)b * NSPLIT + sp) * DK + d];
    out[(size_t)b * DK + d] = s;
}

// ---------------------------------------------------------------------------
extern "C" void kernel_launch(void* const* d_in, const int* in_sizes, int n_in,
                              void* d_out, int out_size, void* d_ws, size_t ws_size,
                              hipStream_t stream)
{
    const float* keys = (const float*)d_in[0];   // (L, B, 128) f32
    const float* q1   = (const float*)d_in[1];   // (1, B, 128) f32
    const float* q2   = (const float*)d_in[2];   // (1, B, 128) f32
    const int*   mask = (const int*)  d_in[3];   // (L, B) i32
    const float* Wk   = (const float*)d_in[4];   // (75, 128) f32
    const float* Wq1  = (const float*)d_in[5];   // (75, 128) f32
    const float* Wq2  = (const float*)d_in[6];   // (75, 128) f32
    const float* Ws   = (const float*)d_in[7];   // (1, 75) f32
    float* out = (float*)d_out;                  // (1, B, 128) f32

    const int B = in_sizes[1] / DK;              // query1 = 1*B*128
    const int L = in_sizes[0] / (B * DK);        // keys   = L*B*128

    float* scores   = (float*)d_ws;                          // B*L floats
    float* partials = scores + (size_t)B * L;                // B*NSPLIT*DK floats

    dim3 g1((L + ROWS_PER_BLOCK - 1) / ROWS_PER_BLOCK, B);
    pairattn_scores_kernel<<<g1, 128, 0, stream>>>(
        keys, q1, q2, mask, Wk, Wq1, Wq2, Ws, scores, L, B);

    pairattn_softmax_kernel<<<B, 256, 0, stream>>>(scores, L, B);

    pairattn_ctx_partial_kernel<<<dim3(B, NSPLIT), 256, 0, stream>>>(
        keys, scores, partials, L, B);

    pairattn_ctx_reduce_kernel<<<B, DK, 0, stream>>>(partials, out, B);
}
